// NERLSTM_CRF_55233279426964
// MI455X (gfx1250) — matrix operand order repaired and checked by reading, not compiled, not run on hardware
//
#include <hip/hip_runtime.h>
#include <hip/hip_bf16.h>

// BiLSTM-CRF on MI455X (gfx1250, wave32).
// Pipeline: gather+pack -> WMMA input GEMM (double-buffered B tiles)
//           -> weight-stationary WMMA recurrence with async global->LDS
//              double-buffered gate slabs -> tag linear -> Viterbi.

typedef __attribute__((ext_vector_type(16))) _Float16 v16h;
typedef __attribute__((ext_vector_type(8)))  float    v8f;
typedef __attribute__((ext_vector_type(4)))  float    v4f;
typedef __attribute__((ext_vector_type(4)))  int      v4i;

#define B_      256
#define S_      512
#define E_      100
#define H_      100
#define T_      9
#define KP      128          // K padded (100 -> 128 = 4 x K32 WMMA steps)
#define NG      400          // 4*H gate width
#define NT      25           // N tiles of 16 (400/16)
#define MT_TOK  8192         // (B*S)/16 token M-tiles
#define TILE_H  (4*32*16)    // halves per 16x128 tile = 2048

// Async global->LDS path (CDNA5 GLOBAL_LOAD_ASYNC_TO_LDS_B128 + ASYNCcnt).
#if defined(__gfx1250__) && \
    __has_builtin(__builtin_amdgcn_global_load_async_to_lds_b128) && \
    __has_builtin(__builtin_amdgcn_s_wait_asynccnt)
#define ASYNC_LDS 1
#else
#define ASYNC_LDS 0
#endif

__device__ __forceinline__ void async_ld16(const float* g, float* l) {
#if ASYNC_LDS
  // Param 1: addrspace(1) int4* (non-const); param 2: addrspace(3) int4*.
  __builtin_amdgcn_global_load_async_to_lds_b128(
      (__attribute__((address_space(1))) v4i*)const_cast<float*>(g),
      (__attribute__((address_space(3))) v4i*)l, 0, 0);
#else
  *(v4f*)l = *(const v4f*)g;   // synchronous fallback
#endif
}
__device__ __forceinline__ void async_wait0() {
#if ASYNC_LDS
  __builtin_amdgcn_s_wait_asynccnt(0);
#endif
}

// ---------------------------------------------------------------------------
// A-matrix (16x32 f16) per-lane layout (interleaved):
//   lane<16 : M=lane,    half h: h<8 -> K=h,   h>=8 -> K=16+(h-8)
//   lane>=16: M=lane-16, half h: h<8 -> K=8+h, h>=8 -> K=24+(h-8)
// B-matrix (32x16 f16) per-lane layout (interleaved, same K order as A):
//   lane<16 : N=lane,    K as A lane<16 ; lane>=16: N=lane-16, K as A lane>=16
// Tiles stored [tile][kb][lane][16 halves] -> one contiguous v16h per lane.
// ---------------------------------------------------------------------------

__global__ void k_embed_gather(const int* __restrict__ x,
                               const float* __restrict__ embed,
                               _Float16* __restrict__ embA) {
  int m = blockIdx.x * blockDim.x + threadIdx.x;   // m = b*512 + s
  if (m >= B_ * S_) return;
  const float* row = embed + (size_t)x[m] * E_;
  int mt = m >> 4, mr = m & 15;
  size_t base = (size_t)mt * TILE_H;
  for (int k = 0; k < KP; ++k) {
    float v = (k < E_) ? row[k] : 0.0f;
    int kb = k >> 5, kk = k & 31;
    int hi = (kk >> 3) & 1;
    int lane = mr + 16 * hi;
    int hidx = (kk & 7) + ((kk >= 16) ? 8 : 0);
    embA[base + ((size_t)kb * 32 + lane) * 16 + hidx] = (_Float16)v;
  }
}

__global__ void k_convert_w(const float* __restrict__ W,
                            _Float16* __restrict__ Wt, int rows, int K) {
  int n = blockIdx.x * blockDim.x + threadIdx.x;
  if (n >= rows) return;
  int ntile = n >> 4, nr = n & 15;
  for (int k = 0; k < KP; ++k) {
    float v = (k < K) ? W[(size_t)n * K + k] : 0.0f;
    int kb = k >> 5, kk = k & 31;
    int lane = nr + 16 * ((kk >> 3) & 1);
    int hidx = (kk & 7) + ((kk >= 16) ? 8 : 0);
    Wt[((size_t)(ntile * 4 + kb) * 32 + lane) * 16 + hidx] = (_Float16)v;
  }
}

// Input GEMM: G[d][s][b][0:400] = emb[m] @ Wih_d^T + b_d  (m = b*512+s).
// One wave per token M-tile, A in registers, B tiles double-buffered so
// global_load_b128s for tile i+1 are in flight under tile i's WMMA chain.
__global__ __launch_bounds__(256)
void k_gemm_in(const _Float16* __restrict__ embA,
               const _Float16* __restrict__ WihT,   // [2*25] B-tiles
               const float* __restrict__ bias_f,
               const float* __restrict__ bias_b,
               float* __restrict__ G) {
  __shared__ float biasS[2 * NG];
  int tid = threadIdx.x;
  for (int i = tid; i < 2 * NG; i += 256)
    biasS[i] = (i < NG) ? bias_f[i] : bias_b[i - NG];
  __syncthreads();

  int lane = tid & 31;
  int wave = tid >> 5;
  int mt = blockIdx.x * 8 + wave;
  if (mt < MT_TOK) {
    v16h a[4];
    const v16h* Ap = (const v16h*)(embA + (size_t)mt * TILE_H);
#pragma unroll
    for (int kb = 0; kb < 4; ++kb) a[kb] = Ap[kb * 32 + lane];

    int ncol = lane & 15;
    int rsel = (lane >> 4) << 3;

    auto loadB = [&](int dn, v16h (&bm)[4]) {
      const v16h* Bp = (const v16h*)(WihT + (size_t)dn * TILE_H);
#pragma unroll
      for (int kb = 0; kb < 4; ++kb) bm[kb] = Bp[kb * 32 + lane];
    };
    auto tileBody = [&](int dn, const v16h (&bm)[4]) {
      int d = dn / NT, nt = dn - d * NT;
      int n = nt * 16 + ncol;
      v8f c;
      float bv = biasS[d * NG + n];
#pragma unroll
      for (int r = 0; r < 8; ++r) c[r] = bv;
#pragma unroll
      for (int kb = 0; kb < 4; ++kb)
        c = __builtin_amdgcn_wmma_f32_16x16x32_f16(
                false, a[kb], false, bm[kb], (short)0, c, false, false);
#pragma unroll
      for (int r = 0; r < 8; ++r) {
        int rowm = mt * 16 + r + rsel;            // tile spans one b, 16 s's
        int bidx = rowm >> 9;
        int sidx = rowm & 511;
        G[(((size_t)d * S_ + sidx) * B_ + bidx) * NG + n] = c[r];
      }
    };

    v16h b0[4], b1[4];
    loadB(0, b0);
    for (int dn = 0; dn < 2 * NT; dn += 2) {       // 50 tiles, 2 per iter
      loadB(dn + 1, b1);
      tileBody(dn, b0);
      loadB((dn + 2 < 2 * NT) ? dn + 2 : (2 * NT - 1), b0);
      tileBody(dn + 1, b1);
    }
  }
}

// Recurrence: block = (direction d, batch-tile bt of 16 rows), 512 steps.
// Whh B-tiles weight-stationary in VGPRs. Gate-input slab G[s] (25.6 KB)
// double-buffered in LDS via async global->LDS, issued one step ahead.
#define SLAB_F   (16 * NG)                 // floats per step slab
#define SLAB_V4  (SLAB_F / 4)              // 1600 x 16B chunks
#define LSTM_SMEM (TILE_H * 2 + SLAB_F * 4 + 16 * H_ * 4 + 2 * SLAB_F * 4)

__global__ __launch_bounds__(256)
void k_lstm(const _Float16* __restrict__ WhhT,    // [2*25] B-tiles
            const float* __restrict__ G,          // [2][S][B][400]
            _Float16* __restrict__ hs) {          // [2][B][S][100] f16
  extern __shared__ __align__(16) char smem[];
  _Float16* hbuf  = (_Float16*)smem;                          // 4 KB
  float*    gates = (float*)(smem + TILE_H * 2);              // 25.6 KB
  float*    cst   = (float*)(smem + TILE_H * 2 + SLAB_F * 4); // 6.4 KB
  float*    gls   = (float*)(smem + TILE_H * 2 + SLAB_F * 4 + 16 * H_ * 4);

  int tid = threadIdx.x;
  int lane = tid & 31;
  int wave = tid >> 5;
  int d = blockIdx.x >> 4;
  int bt = blockIdx.x & 15;

  for (int i = tid; i < TILE_H; i += 256) hbuf[i] = (_Float16)0.0f;
  for (int i = tid; i < 16 * H_; i += 256) cst[i] = 0.0f;

  // Whh tiles -> registers once (live across all 512 steps).
  v16h Breg[4][4];
  int nt0 = wave * 4;
#pragma unroll
  for (int i = 0; i < 4; ++i) {
    int nt = nt0 + i;
    if (nt < NT) {
      const v16h* Bp = (const v16h*)(WhhT + (size_t)(d * NT + nt) * TILE_H);
#pragma unroll
      for (int kb = 0; kb < 4; ++kb) Breg[i][kb] = Bp[kb * 32 + lane];
    }
  }
  int ncol = lane & 15;
  int rsel = (lane >> 4) << 3;
  int erow = tid >> 4;                      // element-wise: row, 16 thr/row
  int ecol = tid & 15;

  const float* Gbase = G + ((size_t)d * S_ * B_ + bt * 16) * NG;
  auto issue_slab = [&](int s, int buf) {   // uniform 7 async ops per wave
    const float* gp = Gbase + (size_t)s * B_ * NG;
    float* lp = gls + buf * SLAB_F;
#pragma unroll
    for (int i = 0; i < 7; ++i) {
      int idx = tid + i * 256;
      idx = (idx < SLAB_V4) ? idx : (SLAB_V4 - 1);
      async_ld16(gp + idx * 4, lp + idx * 4);
    }
  };

  int s0 = d ? (S_ - 1) : 0;
  issue_slab(s0, 0);
  async_wait0();
  __syncthreads();                          // hbuf/cst init + slab 0 ready

  for (int step = 0; step < S_; ++step) {
    int s = d ? (S_ - 1 - step) : step;
    if (step + 1 < S_) {                    // DMA next slab under compute
      int sn = d ? (S_ - 2 - step) : (step + 1);
      issue_slab(sn, (step + 1) & 1);
    }
    const float* Gcur = gls + (step & 1) * SLAB_F;

    v16h a[4];
#pragma unroll
    for (int kb = 0; kb < 4; ++kb)
      a[kb] = *(const v16h*)&hbuf[((size_t)kb * 32 + lane) * 16];
#pragma unroll
    for (int i = 0; i < 4; ++i) {
      int nt = nt0 + i;
      if (nt < NT) {                        // wave-uniform branch
        int n = nt * 16 + ncol;
        v8f c;
#pragma unroll
        for (int r = 0; r < 8; ++r) c[r] = Gcur[(r + rsel) * NG + n];
#pragma unroll
        for (int kb = 0; kb < 4; ++kb)
          c = __builtin_amdgcn_wmma_f32_16x16x32_f16(
                  false, a[kb], false, Breg[i][kb], (short)0, c, false, false);
#pragma unroll
        for (int r = 0; r < 8; ++r) gates[(r + rsel) * NG + n] = c[r];
      }
    }
    __syncthreads();
    // LSTM cell (gate order i,f,g,o): 16 rows x 100 hidden, 16 threads/row
#pragma unroll
    for (int i = 0; i < 7; ++i) {
      int j = ecol + 16 * i;
      if (j < H_) {
        float gi = gates[erow * NG + j];
        float gf = gates[erow * NG + H_ + j];
        float gg = gates[erow * NG + 2 * H_ + j];
        float go = gates[erow * NG + 3 * H_ + j];
        float si = 1.0f / (1.0f + __expf(-gi));
        float sf = 1.0f / (1.0f + __expf(-gf));
        float so = 1.0f / (1.0f + __expf(-go));
        float e2 = __expf(2.0f * gg);
        float tg = (e2 - 1.0f) / (e2 + 1.0f);
        float cn = sf * cst[erow * H_ + j] + si * tg;
        float ec = __expf(2.0f * cn);
        float tc = (ec - 1.0f) / (ec + 1.0f);
        float h = so * tc;
        cst[erow * H_ + j] = cn;
        hs[(((size_t)d * B_ + bt * 16 + erow) * S_ + s) * H_ + j] = (_Float16)h;
        int kb = j >> 5, kk = j & 31;
        int hi = (kk >> 3) & 1;
        int ln = erow + 16 * hi;
        int hx = (kk & 7) + ((kk >= 16) ? 8 : 0);
        hbuf[((size_t)kb * 32 + ln) * 16 + hx] = (_Float16)h;
      }
    }
    async_wait0();                          // next slab landed in LDS
    __syncthreads();                        // publish hbuf/cst + slab
  }
}

// Tag linear: em[b,s,t] = blin[t] + [hf|hb] . Wlin[t]  (N=9 -> VALU)
__global__ void k_linear(const _Float16* __restrict__ hs,
                         const float* __restrict__ Wlin,  // [9][200]
                         const float* __restrict__ blin,
                         float* __restrict__ em) {
  int tid = blockIdx.x * blockDim.x + threadIdx.x;
  if (tid >= B_ * S_) return;
  int b = tid >> 9, s = tid & 511;
  const _Float16* hf = hs + (((size_t)0 * B_ + b) * S_ + s) * H_;
  const _Float16* hb = hs + (((size_t)1 * B_ + b) * S_ + s) * H_;
  float acc[T_];
#pragma unroll
  for (int t = 0; t < T_; ++t) acc[t] = blin[t];
  for (int j = 0; j < H_; ++j) {
    float vf = (float)hf[j];
    float vb = (float)hb[j];
#pragma unroll
    for (int t = 0; t < T_; ++t)
      acc[t] += vf * Wlin[t * 2 * H_ + j] + vb * Wlin[t * 2 * H_ + H_ + j];
  }
#pragma unroll
  for (int t = 0; t < T_; ++t) em[(size_t)tid * T_ + t] = acc[t];
}

// Viterbi: one thread per batch element (mask is all-true at inference).
__global__ void k_viterbi(const float* __restrict__ em,
                          const float* __restrict__ start,
                          const float* __restrict__ trans,
                          const float* __restrict__ endv,
                          unsigned char* __restrict__ bp,
                          float* __restrict__ out) {
  __shared__ float tr[T_ * T_];
  if (threadIdx.x < T_ * T_) tr[threadIdx.x] = trans[threadIdx.x];
  __syncthreads();
  int b = blockIdx.x * blockDim.x + threadIdx.x;
  if (b >= B_) return;
  float sc[T_];
#pragma unroll
  for (int t = 0; t < T_; ++t) sc[t] = start[t] + em[((size_t)b * S_) * T_ + t];
  for (int s = 1; s < S_; ++s) {
    const float* e = em + ((size_t)b * S_ + s) * T_;
    float ns[T_];
#pragma unroll
    for (int cur = 0; cur < T_; ++cur) {
      float best = sc[0] + tr[cur];
      int bi = 0;
#pragma unroll
      for (int p = 1; p < T_; ++p) {
        float v = sc[p] + tr[p * T_ + cur];
        if (v > best) { best = v; bi = p; }      // first-max like jnp.argmax
      }
      ns[cur] = best + e[cur];
      bp[((size_t)s * B_ + b) * T_ + cur] = (unsigned char)bi;
    }
#pragma unroll
    for (int t = 0; t < T_; ++t) sc[t] = ns[t];
  }
  float best = sc[0] + endv[0];
  int last = 0;
#pragma unroll
  for (int t = 1; t < T_; ++t) {
    float v = sc[t] + endv[t];
    if (v > best) { best = v; last = t; }
  }
  out[(size_t)B_ * S_ + b] = best;
  float* paths = out + (size_t)b * S_;
  paths[S_ - 1] = (float)last;
  int cur = last;
  for (int s = S_ - 1; s >= 1; --s) {
    cur = bp[((size_t)s * B_ + b) * T_ + cur];
    paths[s - 1] = (float)cur;
  }
}

extern "C" void kernel_launch(void* const* d_in, const int* in_sizes, int n_in,
                              void* d_out, int out_size, void* d_ws, size_t ws_size,
                              hipStream_t stream) {
  (void)in_sizes; (void)n_in; (void)out_size; (void)ws_size;
  const int*   x     = (const int*)d_in[0];
  /* d_in[1] = mask: all-true at inference */
  const float* embed = (const float*)d_in[2];
  const float* Wih_f = (const float*)d_in[3];
  const float* Whh_f = (const float*)d_in[4];
  const float* b_f   = (const float*)d_in[5];
  const float* Wih_b = (const float*)d_in[6];
  const float* Whh_b = (const float*)d_in[7];
  const float* b_b   = (const float*)d_in[8];
  const float* Wlin  = (const float*)d_in[9];
  const float* blin  = (const float*)d_in[10];
  const float* start = (const float*)d_in[11];
  const float* trans = (const float*)d_in[12];
  const float* endv  = (const float*)d_in[13];

  char* ws = (char*)d_ws;
  const size_t TB = (size_t)TILE_H * sizeof(_Float16);          // 4096 B/tile
  size_t off_embA = 0;                                          // 33.5 MB
  size_t off_Wih  = off_embA + (size_t)MT_TOK * TB;
  size_t off_Whh  = off_Wih  + (size_t)2 * NT * TB;
  size_t off_G    = off_Whh  + (size_t)2 * NT * TB;             // 419 MB
  size_t off_hs   = off_G    + (size_t)2 * S_ * B_ * NG * sizeof(float);
  size_t off_em   = 0;            // reuse embA region (dead after input GEMM)
  size_t off_bp   = off_em + (size_t)B_ * S_ * T_ * sizeof(float);

  _Float16* embA = (_Float16*)(ws + off_embA);
  _Float16* WihT = (_Float16*)(ws + off_Wih);
  _Float16* WhhT = (_Float16*)(ws + off_Whh);
  float*    G    = (float*)(ws + off_G);
  _Float16* hs   = (_Float16*)(ws + off_hs);
  float*    em   = (float*)(ws + off_em);
  unsigned char* bp = (unsigned char*)(ws + off_bp);

  static int smem_set = 0;  // attribute set is idempotent & capture-safe
  if (!smem_set) {
    (void)hipFuncSetAttribute((const void*)k_lstm,
                        hipFuncAttributeMaxDynamicSharedMemorySize, LSTM_SMEM);
    smem_set = 1;
  }

  k_embed_gather<<<(B_ * S_ + 255) / 256, 256, 0, stream>>>(x, embed, embA);
  k_convert_w<<<2, 256, 0, stream>>>(Wih_f, WihT,                NG, E_);
  k_convert_w<<<2, 256, 0, stream>>>(Wih_b, WihT + NT * TILE_H,  NG, E_);
  k_convert_w<<<2, 256, 0, stream>>>(Whh_f, WhhT,                NG, H_);
  k_convert_w<<<2, 256, 0, stream>>>(Whh_b, WhhT + NT * TILE_H,  NG, H_);
  k_gemm_in<<<MT_TOK / 8, 256, 0, stream>>>(embA, WihT, b_f, b_b, G);
  k_lstm<<<32, 256, LSTM_SMEM, stream>>>(WhhT, G, hs);
  k_linear<<<(B_ * S_ + 255) / 256, 256, 0, stream>>>(hs, Wlin, blin, em);
  k_viterbi<<<1, 256, 0, stream>>>(em, start, trans, endv, bp, (float*)d_out);
}
